// NaverNerModel_23484881174629
// MI455X (gfx1250) — matrix operand order repaired and checked by reading, not compiled, not run on hardware
//
#include <hip/hip_runtime.h>
#include <hip/hip_bf16.h>

// ---- problem constants (B,T,V,D,K) = (64, 512, 32000, 768, 29) ----
#define BB   64
#define TT   512
#define DD   768
#define KK   29
#define BT   (BB * TT)          // 32768 rows
#define NEGF (-1.0e30f)

typedef __attribute__((ext_vector_type(16))) __bf16 v16bf;
typedef __attribute__((ext_vector_type(8)))  float  v8f;

// ============================================================================
// Kernel 1: emissions  em[row, k] = dot(emb[x[row]], w[k]) + b[k]
//   row-tile = 16 rows per wave, K padded 29->32 (two 16-wide n-tiles),
//   D reduced in 24 chunks of 32 via v_wmma_f32_16x16x32_bf16.
//   Padded columns (29..31) compute garbage from a clamped (in-bounds) w row
//   and are simply never stored -> no divergence in the hot loop.
// ============================================================================
__global__ __launch_bounds__(256) void emissions_wmma_kernel(
    const int*   __restrict__ x,
    const float* __restrict__ emb,
    const float* __restrict__ w,
    const float* __restrict__ bias,
    float*       __restrict__ em)
{
    const int tid  = threadIdx.x;
    const int lane = tid & 31;
    const int wave = tid >> 5;
    const int row0 = (blockIdx.x * 8 + wave) * 16;

    const int m    = lane & 15;   // sub-row for A gather / N column for B & C
    const int half = lane >> 4;

    // A-side: this lane gathers from row (row0+m) of the embedded sequence
    const int row   = row0 + m;
    const int token = x[row];
    const float* __restrict__ arow = emb + (long)token * DD;

    // B-side: the two output classes this lane covers (n-tile 0 and 1)
    const int  n0   = m;                       // 0..15  (always < 29)
    const int  n1   = 16 + m;                  // 16..31 (valid only when < 29)
    const bool n1ok = (n1 < KK);
    const int  n1c  = n1ok ? n1 : (KK - 1);    // clamp: in-bounds, result dead
    const float* __restrict__ w0 = w + (long)n0  * DD;
    const float* __restrict__ w1 = w + (long)n1c * DD;

    v8f c0 = {};
    v8f c1 = {};

    #pragma unroll 2
    for (int c = 0; c < DD / 32; ++c) {
        const int k0 = c * 32;

        // ---------- A fragment (16x32 bf16): contiguous layout ----------
        // elements 0..15 = K (k0 + half*16 + 0..15)
        float4 a0 = *(const float4*)(arow + k0 + half * 16);
        float4 a1 = *(const float4*)(arow + k0 + half * 16 + 4);
        float4 a2 = *(const float4*)(arow + k0 + half * 16 + 8);
        float4 a3 = *(const float4*)(arow + k0 + half * 16 + 12);
        v16bf a;
        a[0]  = (__bf16)a0.x; a[1]  = (__bf16)a0.y; a[2]  = (__bf16)a0.z; a[3]  = (__bf16)a0.w;
        a[4]  = (__bf16)a1.x; a[5]  = (__bf16)a1.y; a[6]  = (__bf16)a1.z; a[7]  = (__bf16)a1.w;
        a[8]  = (__bf16)a2.x; a[9]  = (__bf16)a2.y; a[10] = (__bf16)a2.z; a[11] = (__bf16)a2.w;
        a[12] = (__bf16)a3.x; a[13] = (__bf16)a3.y; a[14] = (__bf16)a3.z; a[15] = (__bf16)a3.w;

        // ---------- B fragments (32x16 bf16): lanes 0-15 K 0..15, 16-31 K 16..31 ----------
        const int kb = k0 + half * 16;
        float4 b0a = *(const float4*)(w0 + kb);
        float4 b0b = *(const float4*)(w0 + kb + 4);
        float4 b0c = *(const float4*)(w0 + kb + 8);
        float4 b0d = *(const float4*)(w0 + kb + 12);
        v16bf bv0;
        bv0[0]  = (__bf16)b0a.x; bv0[1]  = (__bf16)b0a.y; bv0[2]  = (__bf16)b0a.z; bv0[3]  = (__bf16)b0a.w;
        bv0[4]  = (__bf16)b0b.x; bv0[5]  = (__bf16)b0b.y; bv0[6]  = (__bf16)b0b.z; bv0[7]  = (__bf16)b0b.w;
        bv0[8]  = (__bf16)b0c.x; bv0[9]  = (__bf16)b0c.y; bv0[10] = (__bf16)b0c.z; bv0[11] = (__bf16)b0c.w;
        bv0[12] = (__bf16)b0d.x; bv0[13] = (__bf16)b0d.y; bv0[14] = (__bf16)b0d.z; bv0[15] = (__bf16)b0d.w;

        float4 b1a = *(const float4*)(w1 + kb);
        float4 b1b = *(const float4*)(w1 + kb + 4);
        float4 b1c = *(const float4*)(w1 + kb + 8);
        float4 b1d = *(const float4*)(w1 + kb + 12);
        v16bf bv1;
        bv1[0]  = (__bf16)b1a.x; bv1[1]  = (__bf16)b1a.y; bv1[2]  = (__bf16)b1a.z; bv1[3]  = (__bf16)b1a.w;
        bv1[4]  = (__bf16)b1b.x; bv1[5]  = (__bf16)b1b.y; bv1[6]  = (__bf16)b1b.z; bv1[7]  = (__bf16)b1b.w;
        bv1[8]  = (__bf16)b1c.x; bv1[9]  = (__bf16)b1c.y; bv1[10] = (__bf16)b1c.z; bv1[11] = (__bf16)b1c.w;
        bv1[12] = (__bf16)b1d.x; bv1[13] = (__bf16)b1d.y; bv1[14] = (__bf16)b1d.z; bv1[15] = (__bf16)b1d.w;

        // D = A*B + C  (8 args: neg_a, A, neg_b, B, c_mod, C, reuse_a, reuse_b)
        c0 = __builtin_amdgcn_wmma_f32_16x16x32_bf16(false, a, false, bv0, (short)0, c0, false, false);
        c1 = __builtin_amdgcn_wmma_f32_16x16x32_bf16(false, a, false, bv1, (short)0, c1, false, false);
    }

    // ---------- epilogue: C layout lane -> N = lane&15, M = r + 8*half ----------
    const float bias0 = bias[n0];
    const float bias1 = bias[n1c];
    #pragma unroll
    for (int r = 0; r < 8; ++r) {
        const int orow = row0 + r + 8 * half;
        em[(long)orow * KK + n0] = c0[r] + bias0;
        if (n1ok) em[(long)orow * KK + n1] = c1[r] + bias1;
    }
}

// ============================================================================
// Kernel 2: CRF forward (logsumexp) + numerator + Viterbi + backtrace.
//   One wave32 per batch; one state per lane; backpointers live in LDS.
// ============================================================================
__global__ __launch_bounds__(32) void crf_kernel(
    const float* __restrict__ em_all,
    const int*   __restrict__ tags,
    const float* __restrict__ start_trans,
    const float* __restrict__ end_trans,
    const float* __restrict__ trans,
    float*       __restrict__ nllpart,     // [B] : num - logZ
    float*       __restrict__ out_paths)   // [B*T] floats (d_out + 1)
{
    __shared__ float trans_s[KK * 32];            // row kp, stride 32
    __shared__ float alpha_s[32];
    __shared__ float score_s[32];
    __shared__ float red_s[32];
    __shared__ unsigned char bp_s[(TT - 1) * 32]; // 511 x 32 backpointers
    __shared__ unsigned char path_s[TT];

    const int b    = blockIdx.x;
    const int lane = threadIdx.x;
    const float* __restrict__ em = em_all + (long)b * TT * KK;
    const int*   __restrict__ tg = tags + b * TT;

    // stage transitions: trans_s[kp*32 + kn]; zero the padding columns
    for (int i = lane; i < KK * 32; i += 32) trans_s[i] = 0.0f;
    __syncthreads();
    for (int i = lane; i < KK * KK; i += 32)
        trans_s[(i / KK) * 32 + (i % KK)] = trans[i];
    __syncthreads();

    const bool valid = (lane < KK);
    float alpha = NEGF, score = NEGF;
    if (valid) {
        const float s0 = start_trans[lane] + em[lane];
        alpha = s0;
        score = s0;
    }

    // ---- numerator (gold-path score), fixed-order reduction ----
    float np = 0.0f;
    for (int t = lane; t < TT; t += 32) np += em[t * KK + tg[t]];
    for (int t = lane; t < TT - 1; t += 32) np += trans_s[tg[t] * 32 + tg[t + 1]];
    if (lane == 0) np += start_trans[tg[0]] + end_trans[tg[TT - 1]];
    red_s[lane] = np;
    __syncthreads();
    float num = 0.0f;
    if (lane == 0) {
        for (int i = 0; i < 32; ++i) num += red_s[i];
    }
    __syncthreads();

    // ---- sequential forward + viterbi ----
    for (int t = 1; t < TT; ++t) {
        alpha_s[lane] = alpha;
        score_s[lane] = score;
        __syncthreads();

        float mx = NEGF, vmax = NEGF;
        int bk = 0;
        for (int kp = 0; kp < KK; ++kp) {
            const float tr = trans_s[kp * 32 + lane];
            const float ca = alpha_s[kp] + tr;
            mx = fmaxf(mx, ca);
            const float cv = score_s[kp] + tr;
            if (cv > vmax) { vmax = cv; bk = kp; }   // first-max (jnp.argmax)
        }
        float s = 0.0f;
        for (int kp = 0; kp < KK; ++kp) {
            const float ca = alpha_s[kp] + trans_s[kp * 32 + lane];
            s += __expf(ca - mx);
        }
        const float e = valid ? em[t * KK + lane] : 0.0f;
        alpha = mx + __logf(s) + e;
        score = vmax + e;
        bp_s[(t - 1) * 32 + lane] = (unsigned char)bk;
        __syncthreads();
    }

    // ---- logZ, last-tag argmax, backtrace ----
    red_s[lane]   = valid ? (alpha + end_trans[lane]) : NEGF;
    alpha_s[lane] = valid ? (score + end_trans[lane]) : NEGF;
    __syncthreads();
    if (lane == 0) {
        float mm = NEGF;
        for (int i = 0; i < KK; ++i) mm = fmaxf(mm, red_s[i]);
        float ss = 0.0f;
        for (int i = 0; i < KK; ++i) ss += __expf(red_s[i] - mm);
        const float logZ = mm + __logf(ss);
        nllpart[b] = num - logZ;

        float bm = NEGF; int last = 0;
        for (int i = 0; i < KK; ++i)
            if (alpha_s[i] > bm) { bm = alpha_s[i]; last = i; }
        path_s[TT - 1] = (unsigned char)last;
        int cur = last;
        for (int i = TT - 2; i >= 0; --i) {
            cur = bp_s[i * 32 + cur];
            path_s[i] = (unsigned char)cur;
        }
    }
    __syncthreads();
    for (int t = lane; t < TT; t += 32)
        out_paths[(long)b * TT + t] = (float)path_s[t];
}

// ============================================================================
// Kernel 3: deterministic final NLL = -sum_b (num_b - logZ_b)
// ============================================================================
__global__ void nll_reduce_kernel(const float* __restrict__ nllpart,
                                  float* __restrict__ out)
{
    if (threadIdx.x == 0 && blockIdx.x == 0) {
        float s = 0.0f;
        for (int i = 0; i < BB; ++i) s += nllpart[i];
        out[0] = -s;
    }
}

// ============================================================================
extern "C" void kernel_launch(void* const* d_in, const int* in_sizes, int n_in,
                              void* d_out, int out_size, void* d_ws, size_t ws_size,
                              hipStream_t stream) {
    const int*   x     = (const int*)  d_in[0];
    const int*   tags  = (const int*)  d_in[1];
    const float* emb   = (const float*)d_in[2];
    const float* w     = (const float*)d_in[3];
    const float* bias  = (const float*)d_in[4];
    const float* st    = (const float*)d_in[5];
    const float* et    = (const float*)d_in[6];
    const float* tr    = (const float*)d_in[7];

    float* out = (float*)d_out;       // out[0] = nll, out[1..] = decoded paths (B*T)

    float* em      = (float*)d_ws;            // BT * KK floats (~3.8 MB)
    float* nllpart = em + (size_t)BT * KK;    // 64 floats

    // emissions: 16 rows/wave, 8 waves/block -> 128 rows/block -> 256 blocks
    emissions_wmma_kernel<<<BT / 128, 256, 0, stream>>>(x, emb, w, bias, em);

    // CRF + Viterbi: one wave32 per batch
    crf_kernel<<<BB, 32, 0, stream>>>(em, tags, st, et, tr, nllpart, out + 1);

    // deterministic scalar reduction
    nll_reduce_kernel<<<1, 32, 0, stream>>>(nllpart, out);
}
